// CausalSelfAttention_26285199851655
// MI455X (gfx1250) — matrix operand order repaired and checked by reading, not compiled, not run on hardware
//
#include <hip/hip_runtime.h>
#include <hip/hip_bf16.h>
#include <stdint.h>

// ---------------------------------------------------------------------------
// CDNA5 (gfx1250) causal self-attention:
//   qkv  = bf16(x @ w_attn + b_attn)          (WMMA bf16 GEMM, fused f32->bf16)
//   y    = flash-attention(qkv)  (bf16)        (WMMA QK^T / PV, TDM K-tile stage)
//   out  = y @ w_proj + b_proj   (f32)         (WMMA bf16 GEMM)
// B=2 T=2048 C=1024 H=16 D=64.  M = B*T = 4096.
// ---------------------------------------------------------------------------

typedef __attribute__((ext_vector_type(16))) __bf16 v16bf;
typedef __attribute__((ext_vector_type(8)))  float  v8f;
typedef __attribute__((ext_vector_type(4)))  unsigned int v4u;
typedef __attribute__((ext_vector_type(8)))  unsigned int v8u;

union FragBF {   // 16 bf16 = 32 bytes = one WMMA A/B operand (8 VGPRs)
    uint4 q[2];
    v16bf v;
};

__device__ __forceinline__ unsigned short f2bf(float f) {
    unsigned int u = __float_as_uint(f);
    u += 0x7FFFu + ((u >> 16) & 1u);          // round-to-nearest-even
    return (unsigned short)(u >> 16);
}
__device__ __forceinline__ unsigned int pack2(float a, float b) {
    return (unsigned int)f2bf(a) | ((unsigned int)f2bf(b) << 16);
}

// ---------------------------------------------------------------------------
// GEMM: C[M,N] = A[M,K] @ B[K,N] + bias[N]
// ---------------------------------------------------------------------------
template<bool A_BF16, bool OUT_BF16>
__global__ __launch_bounds__(256)
void gemm_bias_wmma(const void* __restrict__ Ap, const float* __restrict__ Bw,
                    const float* __restrict__ bias, void* __restrict__ Cp,
                    int M, int N, int K)
{
    __shared__ unsigned short Ash[128 * 40];   // [m][k], stride 40 (pad 8)
    __shared__ unsigned short Bsh[128 * 40];   // transposed [n][k], stride 40

    const int tid  = threadIdx.x;
    const int lane = tid & 31;
    const int wid  = tid >> 5;
    const int wm   = wid & 3;       // row strip of 32
    const int wn   = wid >> 2;      // col strip of 64
    const int hi   = lane >> 4;     // lane half
    const int ln   = lane & 15;

    const int rowBase = blockIdx.y * 128;
    const int colBase = blockIdx.x * 128;

    v8f acc[2][4];
    for (int i = 0; i < 2; ++i)
        for (int j = 0; j < 4; ++j)
            for (int v = 0; v < 8; ++v) acc[i][j][v] = 0.f;

    const int nkb = K >> 5;
    for (int kb = 0; kb < nkb; ++kb) {
        const int k0 = kb << 5;
        __syncthreads();
        // ---- stage A tile 128x32 (convert f32->bf16 if needed) ----
        if constexpr (A_BF16) {
            const unsigned short* A = (const unsigned short*)Ap;
            for (int i = 0; i < 4; ++i) {
                int c = i * 256 + tid;                 // chunk of 4 bf16
                int r = c >> 3, c4 = c & 7;
                uint2 u = *(const uint2*)(A + (size_t)(rowBase + r) * K + k0 + c4 * 4);
                *(uint2*)(&Ash[r * 40 + c4 * 4]) = u;
            }
        } else {
            const float* A = (const float*)Ap;
            for (int i = 0; i < 4; ++i) {
                int c = i * 256 + tid;
                int r = c >> 3, c4 = c & 7;
                float4 f = *(const float4*)(A + (size_t)(rowBase + r) * K + k0 + c4 * 4);
                uint2 u; u.x = pack2(f.x, f.y); u.y = pack2(f.z, f.w);
                *(uint2*)(&Ash[r * 40 + c4 * 4]) = u;
            }
        }
        // ---- stage B tile 32x128 transposed ([n][k]) ----
        for (int i = 0; i < 4; ++i) {
            int c = i * 256 + tid;
            int kr = c >> 5, c4 = c & 31;
            float4 f = *(const float4*)(Bw + (size_t)(k0 + kr) * N + colBase + c4 * 4);
            Bsh[(c4 * 4 + 0) * 40 + kr] = f2bf(f.x);
            Bsh[(c4 * 4 + 1) * 40 + kr] = f2bf(f.y);
            Bsh[(c4 * 4 + 2) * 40 + kr] = f2bf(f.z);
            Bsh[(c4 * 4 + 3) * 40 + kr] = f2bf(f.w);
        }
        __syncthreads();

        // ---- fragments: A 16x32 layout (lane-half K-split), B column-per-lane ----
        FragBF a[2], b[4];
        const int kbase = hi * 8;
        for (int mi = 0; mi < 2; ++mi) {
            int row = wm * 32 + mi * 16 + ln;
            a[mi].q[0] = *(const uint4*)(&Ash[row * 40 + kbase]);       // K = kbase..+7
            a[mi].q[1] = *(const uint4*)(&Ash[row * 40 + kbase + 16]);  // K = kbase+16..+23
        }
        const int kb2 = hi * 8;
        for (int ni = 0; ni < 4; ++ni) {
            int nrow = wn * 64 + ni * 16 + ln;
            b[ni].q[0] = *(const uint4*)(&Bsh[nrow * 40 + kb2]);
            b[ni].q[1] = *(const uint4*)(&Bsh[nrow * 40 + kb2 + 16]);
        }
        for (int mi = 0; mi < 2; ++mi)
            for (int ni = 0; ni < 4; ++ni)
                acc[mi][ni] = __builtin_amdgcn_wmma_f32_16x16x32_bf16(
                    false, a[mi].v, false, b[ni].v, (short)0, acc[mi][ni], false, false);
    }

    // ---- epilogue: bias + store (C layout: M = v + 8*hi, N = ln) ----
    for (int mi = 0; mi < 2; ++mi) {
        for (int ni = 0; ni < 4; ++ni) {
            int col = colBase + wn * 64 + ni * 16 + ln;
            float bv = bias[col];
            for (int v = 0; v < 8; ++v) {
                int row = rowBase + wm * 32 + mi * 16 + v + hi * 8;
                float val = acc[mi][ni][v] + bv;
                if constexpr (OUT_BF16)
                    ((unsigned short*)Cp)[(size_t)row * N + col] = f2bf(val);
                else
                    ((float*)Cp)[(size_t)row * N + col] = val;
            }
        }
    }
}

// ---------------------------------------------------------------------------
// Flash attention (causal), bf16 WMMA, f32 online softmax.
// Grid: (T/64, H, B). Block: 128 thr (4 waves); wave w owns 16 query rows.
// K tile (64 rows x 128B, row stride 6144B) is staged into LDS by the Tensor
// Data Mover: one tensor_load_to_lds per key block, issued by wave 0, with
// TDM LDS padding reproducing the 144B padded Ksh row stride.
// ---------------------------------------------------------------------------
__global__ __launch_bounds__(128)
void flash_attn_wmma(const unsigned short* __restrict__ qkv,
                     unsigned short* __restrict__ y)
{
    constexpr int T = 2048, C3 = 3072, Cc = 1024;
    __shared__ unsigned short Ksh[64 * 72];        // [key][d], stride 72 (144B, TDM-padded)
    __shared__ unsigned short Vsh[64 * 72];        // transposed: [d][key]
    __shared__ unsigned short Psh[4 * 16 * 72];    // per-wave P re-layout strip

    const int tid  = threadIdx.x;
    const int lane = tid & 31;
    const int w    = tid >> 5;
    const int hi   = lane >> 4;
    const int ln   = lane & 15;

    const int qt0 = blockIdx.x * 64;
    const int h   = blockIdx.y;
    const int b   = blockIdx.z;
    const size_t rowB = (size_t)b * T;

    // Q fragments (A layout), loaded once
    FragBF qf[2];
    {
        const int kbase = hi * 8;
        const unsigned short* p = qkv + (rowB + qt0 + w * 16 + ln) * C3 + h * 64;
        for (int ks = 0; ks < 2; ++ks) {
            qf[ks].q[0] = *(const uint4*)(p + ks * 32 + kbase);
            qf[ks].q[1] = *(const uint4*)(p + ks * 32 + kbase + 16);
        }
    }

    v8f o[4];
    for (int i = 0; i < 4; ++i)
        for (int v = 0; v < 8; ++v) o[i][v] = 0.f;
    float m_run[8], l_run[8];
    for (int v = 0; v < 8; ++v) { m_run[v] = -__builtin_inff(); l_run[v] = 0.f; }

    const float scale = 0.125f;                    // 1/sqrt(64)
    const int nkb = (int)blockIdx.x + 1;           // causal: key blocks 0..qtile

    // TDM descriptor constants (group1): data_size=8B, pad after 32 DWORDs by
    // 4 DWORDs (128B row -> 144B LDS stride), tile 16x64 (8B units), stride 768.
    const unsigned ldsK = (unsigned)(uintptr_t)&Ksh[0];   // LDS byte offset (low 32 bits)

    for (int kb = 0; kb < nkb; ++kb) {
        const int kb0 = kb * 64;
        __syncthreads();                           // previous tile fully consumed

        // ---- TDM: stage K tile into Ksh ----
        if (w == 0) {
            unsigned long long ga = (unsigned long long)(uintptr_t)
                (qkv + (rowB + kb0) * C3 + Cc + h * 64);
            v4u g0;
            g0.x = 0x1u;                                   // count=1, user mode
            g0.y = ldsK;                                   // lds_addr
            g0.z = (unsigned)(ga & 0xFFFFFFFFu);           // global_addr[31:0]
            g0.w = (unsigned)(ga >> 32) | (2u << 30);      // global_addr[56:32], type=2
            v8u g1;
            g1.s0 = (3u << 16) | (1u << 20) | (4u << 22) | (3u << 25);
            g1.s1 = (16u << 16);                           // tensor_dim0 = 16 (8B units)
            g1.s2 = (64u << 16);                           // tensor_dim1 = 64 rows
            g1.s3 = (16u << 16);                           // tile_dim0 = 16 (8B units)
            g1.s4 = 64u;                                   // tile_dim1 = 64 rows
            g1.s5 = 768u;                                  // tensor_dim0_stride = 6144B/8
            g1.s6 = 0u;
            g1.s7 = 0u;
            asm volatile("tensor_load_to_lds %0, %1" :: "s"(g0), "s"(g1) : "memory");
        }

        // ---- cooperative stage: V transposed ----
        {
            const int r = tid >> 1;
            const int half = tid & 1;
            const size_t gr = rowB + kb0 + r;
            const uint4* vp = (const uint4*)(qkv + gr * C3 + 2 * Cc + h * 64 + half * 32);
            for (int i = 0; i < 4; ++i) {
                union { uint4 q; unsigned short s[8]; } u;
                u.q = vp[i];
                int d0 = half * 32 + i * 8;
                for (int e = 0; e < 8; ++e) Vsh[(d0 + e) * 72 + r] = u.s[e];
            }
        }
        if (w == 0) __builtin_amdgcn_s_wait_tensorcnt(0);
        __syncthreads();

        // ---- S = scale * Q @ K^T with causal mask ----
        float S[4][8];
        for (int kt = 0; kt < 4; ++kt) {
            v8f s;
            for (int v = 0; v < 8; ++v) s[v] = 0.f;
            for (int ks = 0; ks < 2; ++ks) {
                FragBF kf;
                const unsigned short* kp = &Ksh[(kt * 16 + ln) * 72 + ks * 32 + hi * 8];
                kf.q[0] = *(const uint4*)(kp);
                kf.q[1] = *(const uint4*)(kp + 16);
                s = __builtin_amdgcn_wmma_f32_16x16x32_bf16(
                        false, qf[ks].v, false, kf.v, (short)0, s, false, false);
            }
            const int key = kb0 + kt * 16 + ln;
            for (int v = 0; v < 8; ++v) {
                int q = qt0 + w * 16 + v + hi * 8;
                S[kt][v] = (key <= q) ? s[v] * scale : -__builtin_inff();
            }
        }

        // ---- online softmax (row = lane-half x VGPR index; reduce across 16 lanes) ----
        float fac[8];
        for (int v = 0; v < 8; ++v) {
            float mloc = S[0][v];
            for (int kt = 1; kt < 4; ++kt) mloc = fmaxf(mloc, S[kt][v]);
            for (int off = 1; off < 16; off <<= 1)
                mloc = fmaxf(mloc, __shfl_xor(mloc, off, 16));
            float mnew = fmaxf(m_run[v], mloc);
            fac[v] = __expf(m_run[v] - mnew);
            m_run[v] = mnew;
            float sloc = 0.f;
            for (int kt = 0; kt < 4; ++kt) {
                float p = __expf(S[kt][v] - mnew);
                sloc += p;
                Psh[(w * 16 + v + hi * 8) * 72 + kt * 16 + ln] = f2bf(p);
            }
            for (int off = 1; off < 16; off <<= 1)
                sloc += __shfl_xor(sloc, off, 16);
            l_run[v] = l_run[v] * fac[v] + sloc;
        }
        for (int ni = 0; ni < 4; ++ni)
            for (int v = 0; v < 8; ++v)
                o[ni][v] *= fac[v];

        // ---- O += P @ V  (P re-read in A layout from per-wave LDS strip) ----
        FragBF pf[2];
        {
            const unsigned short* pp = &Psh[(w * 16 + ln) * 72];
            const int kbase = hi * 8;
            for (int ks = 0; ks < 2; ++ks) {
                pf[ks].q[0] = *(const uint4*)(pp + ks * 32 + kbase);
                pf[ks].q[1] = *(const uint4*)(pp + ks * 32 + kbase + 16);
            }
        }
        for (int ni = 0; ni < 4; ++ni) {
            for (int ks = 0; ks < 2; ++ks) {
                FragBF vf;
                const unsigned short* vp = &Vsh[(ni * 16 + ln) * 72 + ks * 32 + hi * 8];
                vf.q[0] = *(const uint4*)(vp);
                vf.q[1] = *(const uint4*)(vp + 16);
                o[ni] = __builtin_amdgcn_wmma_f32_16x16x32_bf16(
                            false, pf[ks].v, false, vf.v, (short)0, o[ni], false, false);
            }
        }
    }

    // ---- normalize and store y ----
    for (int v = 0; v < 8; ++v) {
        float inv = 1.f / l_run[v];
        size_t row = rowB + qt0 + w * 16 + v + hi * 8;
        for (int ni = 0; ni < 4; ++ni)
            y[row * Cc + h * 64 + ni * 16 + ln] = f2bf(o[ni][v] * inv);
    }
}

// ---------------------------------------------------------------------------
extern "C" void kernel_launch(void* const* d_in, const int* in_sizes, int n_in,
                              void* d_out, int out_size, void* d_ws, size_t ws_size,
                              hipStream_t stream)
{
    (void)in_sizes; (void)n_in; (void)out_size; (void)ws_size;
    const float* x      = (const float*)d_in[0];
    const float* w_attn = (const float*)d_in[1];
    const float* b_attn = (const float*)d_in[2];
    const float* w_proj = (const float*)d_in[3];
    const float* b_proj = (const float*)d_in[4];
    float* out = (float*)d_out;

    const int M = 4096;                                    // B*T
    unsigned short* qkv = (unsigned short*)d_ws;           // [4096, 3072] bf16  (24 MB)
    unsigned short* y   = qkv + (size_t)M * 3072;          // [4096, 1024] bf16  (8 MB)

    dim3 g1(3072 / 128, M / 128);
    gemm_bias_wmma<false, true><<<g1, 256, 0, stream>>>(x, w_attn, b_attn, qkv, M, 3072, 1024);

    dim3 g2(2048 / 64, 16, 2);
    flash_attn_wmma<<<g2, 128, 0, stream>>>(qkv, y);

    dim3 g3(1024 / 128, M / 128);
    gemm_bias_wmma<true, false><<<g3, 256, 0, stream>>>(y, w_proj, b_proj, out, M, 1024, 1024);
}
